// KDE_23295902613986
// MI455X (gfx1250) — matrix operand order repaired and checked by reading, not compiled, not run on hardware
//
#include <hip/hip_runtime.h>
#include <hip/hip_bf16.h>

typedef __attribute__((ext_vector_type(16))) __bf16 bf16x16;
typedef __attribute__((ext_vector_type(8)))  __bf16 bf16x8;
typedef __attribute__((ext_vector_type(8)))  float  f32x8;
typedef __attribute__((ext_vector_type(4)))  int    v4i;

#define D_DIM 512
#define MTILE 16
#define NTILE 16
#define WAVES 8
#define BLOCK 256
#define KFRAGS 16            // 512 / 32 WMMA k-steps
#define QK     128           // K elements per staged quarter
#define QFRAGS 4             // WMMA k-steps per quarter
#define BROWB  (QK * 2 + 16) // 272 B per staged row (16B pad -> conflict-free ds_load_b128)
#define BUFB   (MTILE * BROWB)
#define LDA    (D_DIM + 8)

#if __has_builtin(__builtin_amdgcn_global_load_async_to_lds_b128)
#define HAVE_ASYNC_LDS 1
#else
#define HAVE_ASYNC_LDS 0
#endif

typedef __attribute__((address_space(1))) v4i* as1_v4i_p;
typedef __attribute__((address_space(3))) v4i* as3_v4i_p;

template <int CNT>
__device__ __forceinline__ void wait_async() {
#if __has_builtin(__builtin_amdgcn_s_wait_asynccnt)
  __builtin_amdgcn_s_wait_asynccnt(CNT);
#else
  asm volatile("s_wait_asynccnt %0" :: "i"(CNT) : "memory");
#endif
}

// Stage one K-quarter (16 rows x 128 K bf16) of the B tile into this wave's
// LDS buffer: 8 async b128 issues (32 lanes x 16B = 512B each).
__device__ __forceinline__ void stage_quarter(const __bf16* __restrict__ gq,
                                              char* __restrict__ lbuf, int lane) {
  const int r = lane & 15;
  const int h = lane >> 4;
  const char* g = (const char*)gq + (size_t)r * (D_DIM * 2);
  char* l = lbuf + r * BROWB;
#pragma unroll
  for (int j = 0; j < 8; ++j) {
    const int c16 = (h + 2 * j) * 16;
#if HAVE_ASYNC_LDS
    __builtin_amdgcn_global_load_async_to_lds_b128(
        (as1_v4i_p)(g + c16), (as3_v4i_p)(l + c16), 0, 0);
#else
    *(bf16x8*)(l + c16) = *(const bf16x8*)(g + c16);
#endif
  }
}

// ---------------------------------------------------------------------------
// Prep: f32 row -> bf16 row + exact f32 squared norm. One block per row.
// ---------------------------------------------------------------------------
__global__ void __launch_bounds__(BLOCK)
kde_prep(const float* __restrict__ src, __bf16* __restrict__ dst,
         float* __restrict__ norms) {
  const int row = blockIdx.x;
  const float* s = src + (size_t)row * D_DIM;
  __bf16* d = dst + (size_t)row * D_DIM;
  float acc = 0.f;
#pragma unroll
  for (int i = 0; i < D_DIM / BLOCK; ++i) {
    const int idx = threadIdx.x + i * BLOCK;
    const float v = s[idx];
    acc += v * v;
    d[idx] = (__bf16)v;
  }
#pragma unroll
  for (int off = 16; off > 0; off >>= 1) acc += __shfl_xor(acc, off, 32);
  __shared__ float red[WAVES];
  if ((threadIdx.x & 31) == 0) red[threadIdx.x >> 5] = acc;
  __syncthreads();
  if (threadIdx.x == 0) {
    float t = 0.f;
#pragma unroll
    for (int w = 0; w < WAVES; ++w) t += red[w];
    norms[row] = t;
  }
}

// ---------------------------------------------------------------------------
// Main: fused bf16-WMMA GEMM + online logsumexp, async-LDS double buffering.
// Block = 8 waves, 16 M-rows. A fragments live in registers for all of K=512.
// Each wave streams N/8 columns; B is staged per-wave into double-buffered
// LDS K-quarters via GLOBAL_LOAD_ASYNC_TO_LDS_B128 (ASYNCcnt), so the WMMA
// chain only waits on ds_loads while a full quarter is always in flight.
// ---------------------------------------------------------------------------
__global__ void __launch_bounds__(BLOCK)
kde_main(const __bf16* __restrict__ Xb, const __bf16* __restrict__ Sb,
         const float* __restrict__ x2, const float* __restrict__ s2,
         const float* __restrict__ scale_p, float* __restrict__ out, int N) {
  extern __shared__ char smem[];
  char* ldsA  = smem;                            // MTILE x LDA bf16 = 16640 B
  char* bbase = smem + MTILE * LDA * 2;          // 8 waves x 2 x BUFB

  __shared__ float wmax_s[WAVES][MTILE];
  __shared__ float wsum_s[WAVES][MTILE];
  __shared__ float sx2[MTILE];

  const int m0   = blockIdx.x * MTILE;
  const int tid  = threadIdx.x;
  const int wv   = tid >> 5;
  const int lane = tid & 31;
  const float scale = *scale_p;

  char* buf0 = bbase + (size_t)wv * 2 * BUFB;
  char* buf1 = buf0 + BUFB;

  // ---- stage A tile (16 x 512 bf16) into LDS
  {
    const __bf16* gsrc = Xb + (size_t)m0 * D_DIM;
#pragma unroll
    for (int p = 0; p < 4; ++p) {
      const int e = (tid + p * BLOCK) * 8;
      const int r = e / D_DIM, c = e % D_DIM;
      *(bf16x8*)(ldsA + (size_t)(r * LDA + c) * 2) =
          *(const bf16x8*)(gsrc + (size_t)r * D_DIM + c);
    }
    if (tid < MTILE) sx2[tid] = x2[m0 + tid];
  }
  __syncthreads();

  // ---- preload all 16 A fragments (K=512) into registers
  const int arow  = lane & 15;
  const int khalf = lane >> 4;
  bf16x16 afrag[KFRAGS];
#pragma unroll
  for (int kk = 0; kk < KFRAGS; ++kk) {
    const int kb = kk * 32;
    const bf16x8 lo = *(const bf16x8*)(ldsA + (size_t)(arow * LDA + kb + khalf * 16) * 2);
    const bf16x8 hi = *(const bf16x8*)(ldsA + (size_t)(arow * LDA + kb + khalf * 16 + 8) * 2);
    afrag[kk] = (bf16x16){lo[0], lo[1], lo[2], lo[3], lo[4], lo[5], lo[6], lo[7],
                          hi[0], hi[1], hi[2], hi[3], hi[4], hi[5], hi[6], hi[7]};
  }

  float xn[8];
#pragma unroll
  for (int r = 0; r < 8; ++r) xn[r] = sx2[khalf * 8 + r];

  float rmax[8], rsum[8];
#pragma unroll
  for (int r = 0; r < 8; ++r) { rmax[r] = -INFINITY; rsum[r] = 0.f; }

  const int chunks = N / (NTILE * WAVES);        // 64 column-tiles per wave
  const int nstart = wv * chunks * NTILE;
  const int bcol   = lane & 15;
  const __bf16* sbase = Sb + (size_t)nstart * D_DIM;

  // prologue: stage chunk 0, quarter 0
  stage_quarter(sbase, buf0, lane);

  for (int ch = 0; ch < chunks; ++ch) {
    const __bf16* stile = sbase + (size_t)ch * NTILE * D_DIM;
    f32x8 acc = (f32x8){0.f, 0.f, 0.f, 0.f, 0.f, 0.f, 0.f, 0.f};

#pragma unroll
    for (int q = 0; q < 4; ++q) {                // st = ch*4+q ; st&1 == q&1
      char* cur = (q & 1) ? buf1 : buf0;
      char* nxt = (q & 1) ? buf0 : buf1;
      const bool more = (q < 3) || (ch + 1 < chunks);
      if (more) {
        const __bf16* gq = (q < 3) ? (stile + (q + 1) * QK)
                                   : (stile + (size_t)NTILE * D_DIM);
        stage_quarter(gq, nxt, lane);
        wait_async<8>();                         // oldest batch (cur) retired
      } else {
        wait_async<0>();
      }

      const char* cb = cur + (lane & 15) * BROWB + khalf * 32;
#pragma unroll
      for (int kk2 = 0; kk2 < QFRAGS; ++kk2) {
        const bf16x16 bfrag = *(const bf16x16*)(cb + kk2 * 64);
        acc = __builtin_amdgcn_wmma_f32_16x16x32_bf16(
            false, afrag[q * QFRAGS + kk2], false, bfrag, (short)0, acc,
            false, false);
      }
    }

    const int nbase = nstart + ch * NTILE;
    const float sn = s2[nbase + bcol];
#pragma unroll
    for (int r = 0; r < 8; ++r) {
      float d2 = xn[r] + sn - 2.0f * acc[r];
      d2 = fmaxf(d2, 0.f);
      const float e  = -scale * d2;
      const float nm = fmaxf(rmax[r], e);
      rsum[r] = rsum[r] * __expf(rmax[r] - nm) + __expf(e - nm);
      rmax[r] = nm;
    }
  }

  // ---- reduce (max,sum) across the 16 lanes sharing the same rows
#pragma unroll
  for (int r = 0; r < 8; ++r) {
#pragma unroll
    for (int off = 8; off > 0; off >>= 1) {      // masks < 16: stays in-half
      const float om = __shfl_xor(rmax[r], off, 32);
      const float os = __shfl_xor(rsum[r], off, 32);
      const float nm = fmaxf(rmax[r], om);
      rsum[r] = rsum[r] * __expf(rmax[r] - nm) + os * __expf(om - nm);
      rmax[r] = nm;
    }
  }
  if ((lane & 15) == 0) {
#pragma unroll
    for (int r = 0; r < 8; ++r) {
      wmax_s[wv][khalf * 8 + r] = rmax[r];
      wsum_s[wv][khalf * 8 + r] = rsum[r];
    }
  }
  __syncthreads();

  if (tid < MTILE) {
    float m = -INFINITY, s = 0.f;
#pragma unroll
    for (int w = 0; w < WAVES; ++w) {
      const float om = wmax_s[w][tid], os = wsum_s[w][tid];
      const float nm = fmaxf(m, om);
      s = s * __expf(m - nm) + os * __expf(om - nm);
      m = nm;
    }
    const float cst = -logf((float)N)
                    + (D_DIM * 0.5f) * logf(scale / 3.14159265358979323846f);
    out[m0 + tid] = logf(s) + m + cst;
  }
}

// ---------------------------------------------------------------------------
extern "C" void kernel_launch(void* const* d_in, const int* in_sizes, int n_in,
                              void* d_out, int out_size, void* d_ws, size_t ws_size,
                              hipStream_t stream) {
  const float* X     = (const float*)d_in[0];
  const float* S     = (const float*)d_in[1];
  const float* scale = (const float*)d_in[2];
  const int M = in_sizes[0] / D_DIM;
  const int N = in_sizes[1] / D_DIM;

  // workspace: Xb[M*D] bf16 | Sb[N*D] bf16 | x2[M] f32 | s2[N] f32
  char* ws = (char*)d_ws;
  __bf16* Xb = (__bf16*)ws;
  __bf16* Sb = (__bf16*)(ws + (size_t)M * D_DIM * sizeof(__bf16));
  float*  x2 = (float*)(ws + ((size_t)M + N) * D_DIM * sizeof(__bf16));
  float*  s2 = x2 + M;

  const size_t dyn_lds = (size_t)MTILE * LDA * 2 + (size_t)WAVES * 2 * BUFB;

  kde_prep<<<M, BLOCK, 0, stream>>>(X, Xb, x2);
  kde_prep<<<N, BLOCK, 0, stream>>>(S, Sb, s2);
  kde_main<<<M / MTILE, BLOCK, dyn_lds, stream>>>(Xb, Sb, x2, s2, scale,
                                                  (float*)d_out, N);
}
